// VQVAE_64665027608797
// MI455X (gfx1250) — matrix operand order repaired and checked by reading, not compiled, not run on hardware
//
#include <hip/hip_runtime.h>
#include <hip/hip_bf16.h>

typedef __attribute__((ext_vector_type(16))) _Float16 v16h;
typedef __attribute__((ext_vector_type(8)))  float    v8f;

#define EPSV 1e-5f

// ---- workspace layout (float offsets) ----
#define WS_STAT1S 0     // 16 ch sums (conv1)
#define WS_STAT1Q 16    // 16 ch sumsq
#define WS_STAT2S 32    // 4
#define WS_STAT2Q 36    // 4
#define WS_STAT3S 40    // 16 (deconv1)
#define WS_STAT3Q 56    // 16
#define WS_VQACC  72
#define WS_RECACC 73
#define WS_A1     80    // 16
#define WS_BB1    96    // 16
#define WS_A2     112   // 4
#define WS_BB2    116   // 4
#define WS_A3     120   // 16
#define WS_BB3    136   // 16
#define WS_Y1     256                    // 64*16*128*128 = 16,777,216 (reused for y3)
#define WS_Y2     (256 + 16777216)       // 64*4*64*64    =  1,048,576
#define WS_DI     (WS_Y2 + 1048576)      // 64*4*64*64    =  1,048,576

static __device__ __forceinline__ v8f wmma_f16(v16h a, v16h b, v8f c) {
  // D = A(16x32 f16) * B(32x16 f16) + C(16x16 f32)
  return __builtin_amdgcn_wmma_f32_16x16x32_f16(false, a, false, b, (short)0, c,
                                                false, false);
}

__global__ void k0_zero(float* ws) {
  if (threadIdx.x < 256) ws[threadIdx.x] = 0.f;
}

// ---------------- conv1: 1->16, k4 s2 p1, GEMM M=16 K=16 N=16/tile -------------
__global__ void k1_conv1(const float* __restrict__ x, const float* __restrict__ w1,
                         const float* __restrict__ b1, float* __restrict__ ws) {
  __shared__ float lsum[16], lsq[16];
  int t = threadIdx.x;
  if (t < 16) { lsum[t] = 0.f; lsq[t] = 0.f; }
  __syncthreads();
  float* y1 = ws + WS_Y1;
  const int lane = t & 31, wid = t >> 5;
  const int m = lane & 15, hi = lane >> 4;
  // A fragment: rows of W1[oc=16][k=ky*4+kx (16)], K padded to 32 with zeros.
  // A/B k order used here (contiguous): lanes0-15 hold K0..15 (halves0-15);
  // lanes16-31 hold K16..31 (=0).
  v16h a = {};
  {
    int kbase = 0;
    for (int j = 0; j < (hi ? 0 : 8); ++j) {
      a[2 * j]     = (_Float16)w1[m * 16 + kbase + 2 * j];
      a[2 * j + 1] = (_Float16)w1[m * 16 + kbase + 2 * j + 1];
    }
  }
  float bias[8];
  for (int v = 0; v < 8; ++v) bias[v] = b1[hi * 8 + v];
  float ls[8], lq[8];
  for (int v = 0; v < 8; ++v) { ls[v] = 0.f; lq[v] = 0.f; }
  const int totalWaves = gridDim.x * (blockDim.x >> 5);
  const int gw = blockIdx.x * (blockDim.x >> 5) + wid;
  const int nTiles = (64 * 128 * 128) / 16;
  for (int tile = gw; tile < nTiles; tile += totalWaves) {
    int p0 = tile * 16;
    int b = p0 >> 14, rem = p0 & 16383;
    int oy = rem >> 7, ox0 = rem & 127;
    // B fragment: col n = lane&15; lanes0-15 hold K0..15, lanes16-31 K16..31(=0)
    v16h bf = {};
    if (hi == 0) {
      int ox = ox0 + m;
      const float* xb = x + (size_t)b * 65536;
      __builtin_prefetch(xb + (2 * oy) * 256 + 2 * ox, 0, 0);
      for (int tt = 0; tt < 16; ++tt) {
        int ky = tt >> 2, kx = tt & 3;
        int iy = 2 * oy + ky - 1, ix = 2 * ox + kx - 1;
        float v = 0.f;
        if (iy >= 0 && iy < 256 && ix >= 0 && ix < 256) v = xb[iy * 256 + ix];
        bf[tt] = (_Float16)v;
      }
    }
    v8f c = {};
    c = wmma_f16(a, bf, c);
    // D layout: lane L: col n=L&15, rows oc = (L>=16)*8 + v
    int n = lane & 15;
    size_t obase = ((size_t)b * 16) * 16384 + (size_t)oy * 128 + (ox0 + n);
    for (int v = 0; v < 8; ++v) {
      float val = c[v] + bias[v];
      y1[obase + ((size_t)(hi * 8 + v) << 14)] = val;
      ls[v] += val;
      lq[v] += val * val;
    }
  }
  // one cross-lane reduction per kernel (lanes 0..15 of each half share oc)
  int n = lane & 15;
  for (int v = 0; v < 8; ++v) {
    float s = ls[v], q = lq[v];
    for (int off = 1; off < 16; off <<= 1) {
      s += __shfl_xor(s, off, 16);
      q += __shfl_xor(q, off, 16);
    }
    if (n == 0) { atomicAdd(&lsum[hi * 8 + v], s); atomicAdd(&lsq[hi * 8 + v], q); }
  }
  __syncthreads();
  if (t < 16) {
    atomicAdd(&ws[WS_STAT1S + t], lsum[t]);
    atomicAdd(&ws[WS_STAT1Q + t], lsq[t]);
  }
}

// ------------- BN finalize: a = g*rsqrt(var+eps), bb = be - m*a -------------
__global__ void k_fin(float* ws, int sOff, int qOff, int aOff, int bbOff,
                      const float* __restrict__ g, const float* __restrict__ be,
                      int C, float invN) {
  int c = threadIdx.x;
  if (c < C) {
    float m = ws[sOff + c] * invN;
    float v = ws[qOff + c] * invN - m * m;
    float a = g[c] * rsqrtf(v + EPSV);
    ws[aOff + c] = a;
    ws[bbOff + c] = be[c] - m * a;
  }
}

// ---------- conv2: 16->4, k4 s2 p1; GEMM M=16(4) K=256 N=16/tile ----------
__global__ void k3_conv2(const float* __restrict__ w2, const float* __restrict__ b2,
                         float* __restrict__ ws) {
  __shared__ float sa1[16], sbb1[16];
  __shared__ float lsum[4], lsq[4];
  int t = threadIdx.x;
  if (t < 16) { sa1[t] = ws[WS_A1 + t]; sbb1[t] = ws[WS_BB1 + t]; }
  if (t < 4) { lsum[t] = 0.f; lsq[t] = 0.f; }
  __syncthreads();
  const float* y1 = ws + WS_Y1;
  float* y2 = ws + WS_Y2;
  const int lane = t & 31, wid = t >> 5;
  const int m = lane & 15, hi = lane >> 4;
  v16h afr[8];
  for (int s = 0; s < 8; ++s) {
    v16h a = {};
    if (m < 4) {
      for (int j = 0; j < 8; ++j) {
        int kl = (hi ? 8 : 0) + (j < 4 ? 2 * j : 16 + 2 * (j - 4));
        int k = 32 * s + kl;                 // k index = ic*16 + ky*4 + kx
        a[2 * j]     = (_Float16)w2[m * 256 + k];
        a[2 * j + 1] = (_Float16)w2[m * 256 + k + 1];
      }
    }
    afr[s] = a;
  }
  float bias[8];
  for (int v = 0; v < 8; ++v) { int oc = hi * 8 + v; bias[v] = (oc < 4) ? b2[oc] : 0.f; }
  float ls[8], lq[8];
  for (int v = 0; v < 8; ++v) { ls[v] = 0.f; lq[v] = 0.f; }
  const int totalWaves = gridDim.x * (blockDim.x >> 5);
  const int gw = blockIdx.x * (blockDim.x >> 5) + wid;
  const int nTiles = (64 * 64 * 64) / 16;
  for (int tile = gw; tile < nTiles; tile += totalWaves) {
    int p0 = tile * 16;
    int b = p0 >> 12, rem = p0 & 4095;
    int oy = rem >> 6, ox0 = rem & 63;
    int ox = ox0 + m;
    v8f c = {};
#pragma unroll
    for (int s = 0; s < 8; ++s) {
      v16h bf = {};
      int kbase = 32 * s + hi * 8;
      for (int tt = 0; tt < 16; ++tt) {
        int k = kbase + (tt < 8 ? tt : tt + 8);
        int ic = k >> 4, tap = k & 15;
        int ky = tap >> 2, kx = tap & 3;
        int iy = 2 * oy + ky - 1, ix = 2 * ox + kx - 1;
        float v = 0.f;
        if (iy >= 0 && iy < 128 && ix >= 0 && ix < 128) {
          float raw = y1[(((size_t)b * 16 + ic) << 14) + iy * 128 + ix];
          v = fmaxf(raw * sa1[ic] + sbb1[ic], 0.f);   // fused BN+ReLU
        }
        bf[tt] = (_Float16)v;
      }
      c = wmma_f16(afr[s], bf, c);
    }
    int n = lane & 15;
    for (int v = 0; v < 8; ++v) {
      int oc = hi * 8 + v;
      float val = c[v] + bias[v];
      if (oc < 4)
        y2[(((size_t)b * 4 + oc) << 12) + oy * 64 + (ox0 + n)] = val;
      ls[v] += val;
      lq[v] += val * val;
    }
  }
  int n = lane & 15;
  for (int v = 0; v < 4; ++v) {           // only oc<4 are real (hi==0 half)
    float s = ls[v], q = lq[v];
    for (int off = 1; off < 16; off <<= 1) {
      s += __shfl_xor(s, off, 16);
      q += __shfl_xor(q, off, 16);
    }
    if (n == 0 && hi == 0) { atomicAdd(&lsum[v], s); atomicAdd(&lsq[v], q); }
  }
  __syncthreads();
  if (t < 4) {
    atomicAdd(&ws[WS_STAT2S + t], lsum[t]);
    atomicAdd(&ws[WS_STAT2Q + t], lsq[t]);
  }
}

// --------- VQ: pre 1x1 conv -> nearest code -> post 1x1 conv + loss ---------
__global__ void k5_vq(const float* __restrict__ emb, const float* __restrict__ wpre,
                      const float* __restrict__ bpre, const float* __restrict__ wpost,
                      const float* __restrict__ bpost, float* __restrict__ ws) {
  __shared__ float se[128];
  __shared__ float sred[8];
  int t = threadIdx.x;
  if (t < 128) se[t] = emb[t];
  __syncthreads();
  const float* y2 = ws + WS_Y2;
  float* di = ws + WS_DI;
  float a2_[4], bb2_[4];
  for (int c = 0; c < 4; ++c) { a2_[c] = ws[WS_A2 + c]; bb2_[c] = ws[WS_BB2 + c]; }
  int p = blockIdx.x * blockDim.x + t;     // 0..262143
  int b = p >> 12, rem = p & 4095;
  size_t base = ((size_t)b * 4) * 4096 + rem;
  float h2[4];
  for (int c = 0; c < 4; ++c) {
    float v = y2[base + ((size_t)c << 12)];
    h2[c] = fmaxf(v * a2_[c] + bb2_[c], 0.f);
  }
  float z0 = bpre[0], z1 = bpre[1];
  for (int c = 0; c < 4; ++c) { z0 += wpre[c] * h2[c]; z1 += wpre[4 + c] * h2[c]; }
  float bestd = 3.4e38f; int bestk = 0;
  for (int k = 0; k < 64; ++k) {
    float d0 = z0 - se[2 * k], d1 = z1 - se[2 * k + 1];
    float d = d0 * d0 + d1 * d1;
    if (d < bestd) { bestd = d; bestk = k; }
  }
  float q0 = se[2 * bestk], q1 = se[2 * bestk + 1];
  float dv0 = q0 - z0, dv1 = q1 - z1;
  float acc = dv0 * dv0 + dv1 * dv1;
  for (int c = 0; c < 4; ++c)
    di[base + ((size_t)c << 12)] = bpost[c] + wpost[2 * c] * q0 + wpost[2 * c + 1] * q1;
  for (int off = 16; off; off >>= 1) acc += __shfl_xor(acc, off, 32);
  if ((t & 31) == 0) sred[t >> 5] = acc;
  __syncthreads();
  if (t == 0) {
    float s = 0.f;
    for (int i = 0; i < 8; ++i) s += sred[i];
    atomicAdd(&ws[WS_VQACC], s);
  }
}

// ----- convT1: 4->16, k4 s2 p1 (gather form); GEMM M=16 K=64 N=16/tile -----
__global__ void k6_deconv1(const float* __restrict__ wd1, const float* __restrict__ bd1,
                           float* __restrict__ ws) {
  __shared__ float lsum[16], lsq[16];
  int t = threadIdx.x;
  if (t < 16) { lsum[t] = 0.f; lsq[t] = 0.f; }
  __syncthreads();
  const float* di = ws + WS_DI;
  float* y3 = ws + WS_Y1;                 // reuse y1 buffer
  const int lane = t & 31, wid = t >> 5;
  const int m = lane & 15, hi = lane >> 4;
  // A[oc][k=ic*16+ky*4+kx] = wd1[ic][oc][ky][kx]   (torch [in,out,kh,kw])
  v16h afr[2];
  for (int s = 0; s < 2; ++s) {
    v16h a = {};
    for (int j = 0; j < 8; ++j) {
      int kl = (hi ? 8 : 0) + (j < 4 ? 2 * j : 16 + 2 * (j - 4));
      int k0 = 32 * s + kl, k1 = k0 + 1;
      a[2 * j]     = (_Float16)wd1[(k0 >> 4) * 256 + m * 16 + (k0 & 15)];
      a[2 * j + 1] = (_Float16)wd1[(k1 >> 4) * 256 + m * 16 + (k1 & 15)];
    }
    afr[s] = a;
  }
  float bias[8];
  for (int v = 0; v < 8; ++v) bias[v] = bd1[hi * 8 + v];
  float ls[8], lq[8];
  for (int v = 0; v < 8; ++v) { ls[v] = 0.f; lq[v] = 0.f; }
  const int totalWaves = gridDim.x * (blockDim.x >> 5);
  const int gw = blockIdx.x * (blockDim.x >> 5) + wid;
  const int nTiles = (64 * 128 * 128) / 16;
  for (int tile = gw; tile < nTiles; tile += totalWaves) {
    int p0 = tile * 16;
    int b = p0 >> 14, rem = p0 & 16383;
    int oy = rem >> 7, ox0 = rem & 127;
    int ox = ox0 + m;
    v8f c = {};
#pragma unroll
    for (int s = 0; s < 2; ++s) {
      v16h bf = {};
      int kbase = 32 * s + hi * 8;
      for (int tt = 0; tt < 16; ++tt) {
        int k = kbase + (tt < 8 ? tt : tt + 8);
        float v = 0.f;
        int ic = k >> 4, ky = (k >> 2) & 3, kx = k & 3;
        int ty = oy + 1 - ky, tx = ox + 1 - kx;   // o = 2*i + k - 1
        if (!(ty & 1) && !(tx & 1)) {
          int iy = ty >> 1, ix = tx >> 1;
          if (iy >= 0 && iy < 64 && ix >= 0 && ix < 64)
            v = di[(((size_t)b * 4 + ic) << 12) + iy * 64 + ix];
        }
        bf[tt] = (_Float16)v;
      }
      c = wmma_f16(afr[s], bf, c);
    }
    int n = lane & 15;
    size_t obase = ((size_t)b * 16) * 16384 + (size_t)oy * 128 + (ox0 + n);
    for (int v = 0; v < 8; ++v) {
      float val = c[v] + bias[v];
      y3[obase + ((size_t)(hi * 8 + v) << 14)] = val;
      ls[v] += val;
      lq[v] += val * val;
    }
  }
  int n = lane & 15;
  for (int v = 0; v < 8; ++v) {
    float s = ls[v], q = lq[v];
    for (int off = 1; off < 16; off <<= 1) {
      s += __shfl_xor(s, off, 16);
      q += __shfl_xor(q, off, 16);
    }
    if (n == 0) { atomicAdd(&lsum[hi * 8 + v], s); atomicAdd(&lsq[hi * 8 + v], q); }
  }
  __syncthreads();
  if (t < 16) {
    atomicAdd(&ws[WS_STAT3S + t], lsum[t]);
    atomicAdd(&ws[WS_STAT3Q + t], lsq[t]);
  }
}

// -------- convT2: 16->1 + tanh + recon-loss accumulation (scalar, BW-bound) ----
__global__ void k8_deconv2(const float* __restrict__ x, const float* __restrict__ wd2,
                           const float* __restrict__ bd2, float* __restrict__ ws,
                           float* __restrict__ out) {
  __shared__ float sa[16], sb[16], sw[256];
  __shared__ float sred[8];
  int t = threadIdx.x;
  if (t < 16) { sa[t] = ws[WS_A3 + t]; sb[t] = ws[WS_BB3 + t]; }
  if (t < 256) sw[t] = wd2[t];
  __syncthreads();
  const float* y3 = ws + WS_Y1;
  size_t p = (size_t)blockIdx.x * blockDim.x + t;   // 0..4194303
  int b = (int)(p >> 16), rem = (int)(p & 65535);
  int oy = rem >> 8, ox = rem & 255;
  int kys[2], iys[2], kxs[2], ixs[2];
  if (oy & 1) { kys[0] = 0; iys[0] = (oy + 1) >> 1; kys[1] = 2; iys[1] = (oy - 1) >> 1; }
  else        { kys[0] = 1; iys[0] = oy >> 1;       kys[1] = 3; iys[1] = (oy >> 1) - 1; }
  if (ox & 1) { kxs[0] = 0; ixs[0] = (ox + 1) >> 1; kxs[1] = 2; ixs[1] = (ox - 1) >> 1; }
  else        { kxs[0] = 1; ixs[0] = ox >> 1;       kxs[1] = 3; ixs[1] = (ox >> 1) - 1; }
  float acc = bd2[0];
  for (int ic = 0; ic < 16; ++ic) {
    const float* yb = y3 + (((size_t)b * 16 + ic) << 14);
    float a = sa[ic], bb = sb[ic];
    for (int u = 0; u < 2; ++u) {
      int iy = iys[u];
      if (iy < 0 || iy >= 128) continue;
      for (int w = 0; w < 2; ++w) {
        int ix = ixs[w];
        if (ix < 0 || ix >= 128) continue;
        float h = fmaxf(yb[iy * 128 + ix] * a + bb, 0.f);
        acc += h * sw[ic * 16 + kys[u] * 4 + kxs[w]];
      }
    }
  }
  float o = tanhf(acc);
  out[p] = o;
  float d = x[p] - o;
  float rl = d * d;
  for (int off = 16; off; off >>= 1) rl += __shfl_xor(rl, off, 32);
  if ((t & 31) == 0) sred[t >> 5] = rl;
  __syncthreads();
  if (t == 0) {
    float s = 0.f;
    for (int i = 0; i < 8; ++i) s += sred[i];
    atomicAdd(&ws[WS_RECACC], s);
  }
}

__global__ void k9_loss(const float* __restrict__ ws, float* __restrict__ out) {
  // quantize_loss = (1 + 0.2)*mean((q-z)^2) + recon_mse
  out[4194304] = ws[WS_VQACC] * (1.2f / 524288.f) + ws[WS_RECACC] * (1.f / 4194304.f);
}

extern "C" void kernel_launch(void* const* d_in, const int* in_sizes, int n_in,
                              void* d_out, int out_size, void* d_ws, size_t ws_size,
                              hipStream_t stream) {
  (void)in_sizes; (void)n_in; (void)out_size; (void)ws_size;
  const float* x     = (const float*)d_in[0];
  const float* w1    = (const float*)d_in[1];
  const float* b1    = (const float*)d_in[2];
  const float* g1    = (const float*)d_in[3];
  const float* be1   = (const float*)d_in[4];
  const float* w2    = (const float*)d_in[5];
  const float* b2    = (const float*)d_in[6];
  const float* g2    = (const float*)d_in[7];
  const float* be2   = (const float*)d_in[8];
  const float* wpre  = (const float*)d_in[9];
  const float* bpre  = (const float*)d_in[10];
  const float* emb   = (const float*)d_in[11];
  const float* wpost = (const float*)d_in[12];
  const float* bpost = (const float*)d_in[13];
  const float* wd1   = (const float*)d_in[14];
  const float* bd1   = (const float*)d_in[15];
  const float* g3    = (const float*)d_in[16];
  const float* be3   = (const float*)d_in[17];
  const float* wd2   = (const float*)d_in[18];
  const float* bd2   = (const float*)d_in[19];
  float* ws  = (float*)d_ws;
  float* out = (float*)d_out;

  k0_zero<<<1, 256, 0, stream>>>(ws);
  k1_conv1<<<1024, 128, 0, stream>>>(x, w1, b1, ws);
  k_fin<<<1, 32, 0, stream>>>(ws, WS_STAT1S, WS_STAT1Q, WS_A1, WS_BB1, g1, be1, 16,
                              1.f / 1048576.f);
  k3_conv2<<<512, 128, 0, stream>>>(w2, b2, ws);
  k_fin<<<1, 32, 0, stream>>>(ws, WS_STAT2S, WS_STAT2Q, WS_A2, WS_BB2, g2, be2, 4,
                              1.f / 262144.f);
  k5_vq<<<1024, 256, 0, stream>>>(emb, wpre, bpre, wpost, bpost, ws);
  k6_deconv1<<<1024, 128, 0, stream>>>(wd1, bd1, ws);
  k_fin<<<1, 32, 0, stream>>>(ws, WS_STAT3S, WS_STAT3Q, WS_A3, WS_BB3, g3, be3, 16,
                              1.f / 1048576.f);
  k8_deconv2<<<16384, 256, 0, stream>>>(x, wd2, bd2, ws, out);
  k9_loss<<<1, 1, 0, stream>>>(ws, out);
}
